// HexPlane_44272522887577
// MI455X (gfx1250) — matrix operand order repaired and checked by reading, not compiled, not run on hardware
//
#include <hip/hip_runtime.h>
#include <hip/hip_bf16.h>

typedef __attribute__((ext_vector_type(16))) _Float16 v16h;
typedef __attribute__((ext_vector_type(8)))  _Float16 v8h;
typedef __attribute__((ext_vector_type(8)))  float    v8f;

#define THREADS 128          // 4 waves
#define NPTS_BLOCK 128       // 1 point / thread
#define TILES 8              // 16-point WMMA tiles per block
#define LKP 96               // label GEMM K padded (72 -> 96)
#define DKP 32               // deform GEMM K padded (24 -> 32)
#define DOUT 112             // deform output dim (16 labels * 7)
#define GRID_L 128
#define GRID_T 64

__device__ __forceinline__ int iclamp(int v, int lo, int hi) {
    return v < lo ? lo : (v > hi ? hi : v);
}

// A fragment (16-bit 16x32): lane<16 -> K[0..15]; lane>=16 -> K[16..31]
__device__ __forceinline__ v16h load_fragA(const _Float16* rowp, int hi_half) {
    return *(const v16h*)(rowp + (hi_half ? 16 : 0));
}
// B fragment (32x16): lane<16 -> K[0..15]; lane>=16 -> K[16..31] of its column
__device__ __forceinline__ v16h load_fragB(const _Float16* colp, int hi_half) {
    return *(const v16h*)(colp + (hi_half ? 16 : 0));
}

__global__ __launch_bounds__(THREADS)
void hexplane_wmma_kernel(const float* __restrict__ pts_in,
                          const float* __restrict__ rot_in,
                          const float* __restrict__ tsrc,
                          const float* __restrict__ ttgt,
                          const float* __restrict__ label_planes,
                          const float* __restrict__ deform_plane,
                          const float* __restrict__ W_label,
                          const float* __restrict__ W_deform,
                          float* __restrict__ out, int n)
{
    __shared__ __align__(32) _Float16 sWl[16 * LKP];          // B for label GEMM (cols contiguous)
    __shared__ __align__(32) _Float16 sWd[DOUT * DKP];        // B for deform GEMM
    __shared__ __align__(32) _Float16 sFeats[TILES * 16 * LKP]; // A: [tile][row][K]
    __shared__ __align__(32) _Float16 sTf[TILES * 16 * DKP];    // A: [tile][row][K]
    __shared__ __align__(32) float    sLF[TILES * 16 * 16];     // label logits [tile][row][16]
    __shared__ __align__(32) _Float16 sDfm[TILES * 16 * DOUT];  // deform out  [tile][row][112]

    const int tid  = threadIdx.x;
    const int lane = tid & 31;
    const int wv   = tid >> 5;
    const int tg   = tid >> 4;          // tile 0..7 (point-ownership view)
    const int row  = tid & 15;
    const int pg   = blockIdx.x * NPTS_BLOCK + tid;
    const int p    = (pg < n) ? pg : (n - 1);   // clamp so loads stay in range, keep EXEC full

    // ---- convert weights to f16 B layout in LDS (block-cooperative) ----
    for (int i = tid; i < 16 * LKP; i += THREADS) {
        int k = i % LKP, r = i / LKP;
        sWl[i] = (k < 72) ? (_Float16)W_label[r * 72 + k] : (_Float16)0.f;
    }
    for (int i = tid; i < DOUT * DKP; i += THREADS) {
        int k = i % DKP, r = i / DKP;
        sWd[i] = (k < 24) ? (_Float16)W_deform[r * 24 + k] : (_Float16)0.f;
    }

    // ---- per-point inputs ----
    const float px = pts_in[p * 3 + 0], py = pts_in[p * 3 + 1], pz = pts_in[p * 3 + 2];
    const float rw = rot_in[p * 4 + 0], rx = rot_in[p * 4 + 1];
    const float ry = rot_in[p * 4 + 2], rz = rot_in[p * 4 + 3];
    const float ts = tsrc[p], tt = ttgt[p];

    // ---- time-plane bilinear features -> A matrix for deform GEMM ----
    {
        float x = (ts + 1.f) * 0.5f * (GRID_T - 1);
        float y = (tt + 1.f) * 0.5f * (GRID_T - 1);
        float x0f = floorf(x), y0f = floorf(y);
        float wx = x - x0f, wy = y - y0f;
        int x0 = iclamp((int)x0f, 0, GRID_T - 1);
        int x1 = iclamp(x0 + 1, 0, GRID_T - 1);
        int y0 = iclamp((int)y0f, 0, GRID_T - 1);
        int y1 = iclamp(y0 + 1, 0, GRID_T - 1);
        float w00 = (1.f - wx) * (1.f - wy), w01 = wx * (1.f - wy);
        float w10 = (1.f - wx) * wy,         w11 = wx * wy;
        int i00 = y0 * GRID_T + x0, i01 = y0 * GRID_T + x1;
        int i10 = y1 * GRID_T + x0, i11 = y1 * GRID_T + x1;
        _Float16* dst = &sTf[(tg * 16 + row) * DKP];
        #pragma unroll
        for (int c = 0; c < 24; ++c) {
            const float* pl = deform_plane + c * (GRID_T * GRID_T);
            float v = pl[i00] * w00 + pl[i01] * w01 + pl[i10] * w10 + pl[i11] * w11;
            dst[c] = (_Float16)v;
        }
        #pragma unroll
        for (int c = 24; c < DKP; ++c) dst[c] = (_Float16)0.f;
    }

    // ---- label-plane bilinear features -> A matrix for label GEMM ----
    {
        // pts_n = (pts - 1)*(2/(-2)) - 1 = -pts
        const float pn[3] = {-px, -py, -pz};
        _Float16* dst = &sFeats[(tg * 16 + row) * LKP];
        #pragma unroll
        for (int plidx = 0; plidx < 3; ++plidx) {
            float cx = pn[(plidx == 2) ? 1 : 0];
            float cy = pn[(plidx == 0) ? 1 : 2];
            float x = (cx + 1.f) * 0.5f * (GRID_L - 1);
            float y = (cy + 1.f) * 0.5f * (GRID_L - 1);
            float x0f = floorf(x), y0f = floorf(y);
            float wx = x - x0f, wy = y - y0f;
            int x0 = iclamp((int)x0f, 0, GRID_L - 1);
            int x1 = iclamp(x0 + 1, 0, GRID_L - 1);
            int y0 = iclamp((int)y0f, 0, GRID_L - 1);
            int y1 = iclamp(y0 + 1, 0, GRID_L - 1);
            float w00 = (1.f - wx) * (1.f - wy), w01 = wx * (1.f - wy);
            float w10 = (1.f - wx) * wy,         w11 = wx * wy;
            int i00 = y0 * GRID_L + x0, i01 = y0 * GRID_L + x1;
            int i10 = y1 * GRID_L + x0, i11 = y1 * GRID_L + x1;
            const float* base = label_planes + plidx * 24 * (GRID_L * GRID_L);
            #pragma unroll
            for (int c = 0; c < 24; ++c) {
                const float* pl = base + c * (GRID_L * GRID_L);
                float v = pl[i00] * w00 + pl[i01] * w01 + pl[i10] * w10 + pl[i11] * w11;
                dst[plidx * 24 + c] = (_Float16)v;
            }
        }
        #pragma unroll
        for (int k = 72; k < LKP; ++k) dst[k] = (_Float16)0.f;
    }

    __syncthreads();

    // ---- WMMA phase: each wave does 2 M-tiles; 3 + 7 wmma per tile ----
    {
        const int hi    = (lane & 16);          // lane>=16?
        const int colB  = lane & 15;            // B column / A row / D column
        const int rbase = hi ? 8 : 0;           // D row base
        #pragma unroll
        for (int t = 0; t < 2; ++t) {
            const int tgw = wv * 2 + t;
            // label GEMM: [16 x 96] x [96 x 16], K in 3 chunks of 32
            v8f acc = {};
            #pragma unroll
            for (int kc = 0; kc < 3; ++kc) {
                v16h a = load_fragA(&sFeats[(tgw * 16 + colB) * LKP + kc * 32], hi);
                v16h b = load_fragB(&sWl[colB * LKP + kc * 32], hi);
                acc = __builtin_amdgcn_wmma_f32_16x16x32_f16(false, a, false, b,
                                                             (short)0, acc, false, false);
            }
            #pragma unroll
            for (int r = 0; r < 8; ++r)
                sLF[(tgw * 16 + rbase + r) * 16 + colB] = acc[r];

            // deform GEMM: [16 x 32] x [32 x 112], 7 N-tiles
            v16h at = load_fragA(&sTf[(tgw * 16 + colB) * DKP], hi);
            #pragma unroll
            for (int j = 0; j < 7; ++j) {
                v16h b = load_fragB(&sWd[(j * 16 + colB) * DKP], hi);
                v8f acc2 = {};
                acc2 = __builtin_amdgcn_wmma_f32_16x16x32_f16(false, at, false, b,
                                                              (short)0, acc2, false, false);
                #pragma unroll
                for (int r = 0; r < 8; ++r)
                    sDfm[(tgw * 16 + rbase + r) * DOUT + j * 16 + colB] = (_Float16)acc2[r];
            }
        }
    }

    __syncthreads();

    // ---- per-point epilogue ----
    float lf[16];
    {
        const float* lfp = &sLF[(tg * 16 + row) * 16];
        float m = -1e30f;
        #pragma unroll
        for (int k = 0; k < 16; ++k) { lf[k] = lfp[k]; m = fmaxf(m, lf[k]); }
        float s = 0.f;
        #pragma unroll
        for (int k = 0; k < 16; ++k) { lf[k] = __expf((lf[k] - m) * 10.0f); s += lf[k]; }
        float inv = 1.f / s;
        #pragma unroll
        for (int k = 0; k < 16; ++k) lf[k] *= inv;
    }

    float qcw = 0.f, qcx = 0.f, qcy = 0.f, qcz = 0.f;
    float tcx = 0.f, tcy = 0.f, tcz = 0.f;
    {
        const _Float16* dp = &sDfm[(tg * 16 + row) * DOUT];
        #pragma unroll
        for (int k = 0; k < 16; ++k) {
            float qw = (float)dp[k * 7 + 0], qx = (float)dp[k * 7 + 1];
            float qy = (float)dp[k * 7 + 2], qz = (float)dp[k * 7 + 3];
            float tx = (float)dp[k * 7 + 4], ty = (float)dp[k * 7 + 5];
            float tz = (float)dp[k * 7 + 6];
            float nrm = sqrtf(qw * qw + qx * qx + qy * qy + qz * qz);
            float qi  = 1.f / fmaxf(nrm, 1e-12f);
            float l = lf[k];
            float lq = l * qi;
            qcw += lq * qw; qcx += lq * qx; qcy += lq * qy; qcz += lq * qz;
            tcx += l * tx;  tcy += l * ty;  tcz += l * tz;
        }
    }

    // quat_apply(qc, pts): qc * (0,p) * conj(qc), take xyz
    const float aw = qcw, ax = qcx, ay = qcy, az = qcz;
    float t1w = -ax * px - ay * py - az * pz;
    float t1x =  aw * px + ay * pz - az * py;
    float t1y =  aw * py - ax * pz + az * px;
    float t1z =  aw * pz + ax * py - ay * px;
    float ox = -t1w * ax + t1x * aw - t1y * az + t1z * ay;
    float oy = -t1w * ay + t1x * az + t1y * aw - t1z * ax;
    float oz = -t1w * az - t1x * ay + t1y * ax + t1z * aw;

    // quat_multiply(qc, rot) with non-negative real part
    float mw = aw * rw - ax * rx - ay * ry - az * rz;
    float mx = aw * rx + ax * rw + ay * rz - az * ry;
    float my = aw * ry - ax * rz + ay * rw + az * rx;
    float mz = aw * rz + ax * ry - ay * rx + az * rw;
    if (mw < 0.f) { mw = -mw; mx = -mx; my = -my; mz = -mz; }

    if (pg < n) {
        out[pg * 3 + 0] = ox + tcx;
        out[pg * 3 + 1] = oy + tcy;
        out[pg * 3 + 2] = oz + tcz;
        float* rout = out + (size_t)n * 3;
        rout[pg * 4 + 0] = mw; rout[pg * 4 + 1] = mx;
        rout[pg * 4 + 2] = my; rout[pg * 4 + 3] = mz;
        float* lout = out + (size_t)n * 7;
        #pragma unroll
        for (int k = 0; k < 16; ++k) lout[pg * 16 + k] = lf[k];
    }
}

extern "C" void kernel_launch(void* const* d_in, const int* in_sizes, int n_in,
                              void* d_out, int out_size, void* d_ws, size_t ws_size,
                              hipStream_t stream) {
    const float* pts = (const float*)d_in[0];
    const float* rot = (const float*)d_in[1];
    const float* ts  = (const float*)d_in[2];
    const float* tt  = (const float*)d_in[3];
    const float* lp  = (const float*)d_in[4];
    const float* dp  = (const float*)d_in[5];
    const float* wl  = (const float*)d_in[6];
    const float* wd  = (const float*)d_in[7];
    const int n = in_sizes[0] / 3;
    const int blocks = (n + NPTS_BLOCK - 1) / NPTS_BLOCK;
    hexplane_wmma_kernel<<<blocks, THREADS, 0, stream>>>(
        pts, rot, ts, tt, lp, dp, wl, wd, (float*)d_out, n);
}
